// MANNCell_90555090468991
// MI455X (gfx1250) — compile-verified, matrix operand order checked by reading, not run on hardware
//
#include <hip/hip_runtime.h>
#include <math.h>

// MANNCell forward for MI455X (gfx1250, wave32, WMMA).
// B=128, M=2048, I=256, H=512, HEAD=64, AUX_H=130, memcnt==M.
// Output = [h_1 (128x512) | read_head (128x130)] fp32.

#define BB 128
#define MM 2048
#define I_DIM 256
#define H_DIM 512
#define AUXH 130

typedef __attribute__((ext_vector_type(16))) __bf16 v16bf;
typedef __attribute__((ext_vector_type(8)))  float  v8f;

// 16-bit A/B fragment K index for element e (0..15) and lane-half (0/1).
// Per ISA 7.12.2: lanes0-15 VGPR0..3 = K0..7, VGPR4..7 = K16..23; lanes16-31 +8.
__device__ __forceinline__ int kmap(int e, int half) {
  return ((e < 8) ? e : e + 8) + half * 8;
}
__device__ __forceinline__ float sigf(float x) { return 1.0f / (1.0f + expf(-x)); }

// ---------------------------------------------------------------------------
// Prep: convert fp32 weight [Dk,Dn] into bf16 B-fragment order
// dst[((kb*(Dn/16)+nt)*32 + lane)*16 + e] = src[(kb*32 + kmap(e,lane>>4))*Dn + nt*16 + (lane&15)]
// ---------------------------------------------------------------------------
__global__ void prep_bfrag(const float* __restrict__ src, __bf16* __restrict__ dst,
                           int Dk, int Dn) {
  int idx = blockIdx.x * blockDim.x + threadIdx.x;
  int total = (Dk >> 5) * (Dn >> 4) * 512;
  if (idx >= total) return;
  int e = idx & 15;
  int lane = (idx >> 4) & 31;
  int frag = idx >> 9;
  int ntiles = Dn >> 4;
  int nt = frag % ntiles;
  int kb = frag / ntiles;
  int k = kb * 32 + kmap(e, lane >> 4);
  int n = nt * 16 + (lane & 15);
  dst[idx] = (__bf16)src[(size_t)k * Dn + n];
}

// Prep: A-fragment order for concat([src0 (128 x split) | src1 (128 x (Dk-split))])
// dst[((mt*(Dk/32)+kb)*32 + lane)*16 + e] = X[mt*16 + (lane&15)][kb*32 + kmap(e,lane>>4)]
__global__ void prep_afrag(const float* __restrict__ src0, const float* __restrict__ src1,
                           __bf16* __restrict__ dst, int Dk, int split) {
  int idx = blockIdx.x * blockDim.x + threadIdx.x;
  int total = (BB / 16) * (Dk >> 5) * 512;
  if (idx >= total) return;
  int e = idx & 15;
  int lane = (idx >> 4) & 31;
  int frag = idx >> 9;
  int kbs = Dk >> 5;
  int kb = frag % kbs;
  int mt = frag / kbs;
  int m = mt * 16 + (lane & 15);
  int k = kb * 32 + kmap(e, lane >> 4);
  float v = (k < split) ? src0[(size_t)m * split + k]
                        : src1[(size_t)m * (Dk - split) + (k - split)];
  dst[idx] = (__bf16)v;
}

// ---------------------------------------------------------------------------
// Aux GRU: read_head = GRU([input_|h_0] (128x768), aux_h_0 (128x130)). Tiny.
// ---------------------------------------------------------------------------
__global__ void aux_gru_kernel(const float* __restrict__ input_,
                               const float* __restrict__ h0,
                               const float* __restrict__ auxh0,
                               const float* __restrict__ wih,   // [768,390]
                               const float* __restrict__ whh,   // [130,390]
                               const float* __restrict__ bih,   // [390]
                               const float* __restrict__ bhh,   // [390]
                               float* __restrict__ read_head) { // [128,130]
  __shared__ float xs[768];
  __shared__ float hs[AUXH];
  __shared__ float wi[390];
  __shared__ float wh[390];
  int b = blockIdx.x, t = threadIdx.x;
  for (int k = t; k < I_DIM; k += blockDim.x) xs[k] = input_[b * I_DIM + k];
  for (int k = t; k < H_DIM; k += blockDim.x) xs[I_DIM + k] = h0[b * H_DIM + k];
  for (int k = t; k < AUXH; k += blockDim.x) hs[k] = auxh0[b * AUXH + k];
  __syncthreads();
  if (t < 390) {
    float a = bih[t];
    for (int k = 0; k < 768; ++k) a += xs[k] * wih[k * 390 + t];
    wi[t] = a;
    float c = bhh[t];
    for (int k = 0; k < AUXH; ++k) c += hs[k] * whh[k * 390 + t];
    wh[t] = c;
  }
  __syncthreads();
  if (t < AUXH) {
    float r = sigf(wi[t] + wh[t]);
    float z = sigf(wi[130 + t] + wh[130 + t]);
    float n = tanhf(wi[260 + t] + r * wh[260 + t]);
    read_head[b * AUXH + t] = (1.f - z) * n + z * hs[t];
  }
}

// ---------------------------------------------------------------------------
// Fused per-slot FC head + score + gumbel:
//   hid = relu(mem[b,m,:] @ w1 + b1)        (WMMA bf16, K=D, N=32)
//   out = sigmoid(hid @ w2 + b2)            (WMMA bf16, K=32, N=64)
//   sg[b,m] = sum_n out[n]*rh[n] + rh[64]*sigmoid(last_use) + gumbel(u)
// One wave = 16 slots. Block = 8 waves = 128 slots. Grid (B, M/128).
// HBM-bound: dominated by streaming mem (256MB / 512MB).
// ---------------------------------------------------------------------------
template <int D>
__global__ void __launch_bounds__(256) fc_score_kernel(
    const float* __restrict__ mem, const float* __restrict__ last_use,
    const float* __restrict__ u,
    const __bf16* __restrict__ w1f, const float* __restrict__ b1,
    const __bf16* __restrict__ w2f, const float* __restrict__ b2,
    const float* __restrict__ read_head, int rh_off,
    float* __restrict__ sg_out) {
  __shared__ __bf16 hid[8][16][32];
  int tid = threadIdx.x;
  int w = tid >> 5, lane = tid & 31;
  int half = lane >> 4, lr = lane & 15;
  int b = blockIdx.x;
  int m0 = blockIdx.y * 128 + w * 16;
  const float* row = mem + ((size_t)b * MM + (m0 + lr)) * D;

  constexpr int KB = D / 32;
  v8f acc0 = {};
  v8f acc1 = {};
#pragma unroll
  for (int kb = 0; kb < KB; ++kb) {
    int k0 = kb * 32 + half * 8;
    float4 q0 = *(const float4*)(row + k0);
    float4 q1 = *(const float4*)(row + k0 + 4);
    float4 q2 = *(const float4*)(row + k0 + 16);
    float4 q3 = *(const float4*)(row + k0 + 20);
    v16bf a;
    a[0] = (__bf16)q0.x;  a[1] = (__bf16)q0.y;  a[2] = (__bf16)q0.z;  a[3] = (__bf16)q0.w;
    a[4] = (__bf16)q1.x;  a[5] = (__bf16)q1.y;  a[6] = (__bf16)q1.z;  a[7] = (__bf16)q1.w;
    a[8] = (__bf16)q2.x;  a[9] = (__bf16)q2.y;  a[10] = (__bf16)q2.z; a[11] = (__bf16)q2.w;
    a[12] = (__bf16)q3.x; a[13] = (__bf16)q3.y; a[14] = (__bf16)q3.z; a[15] = (__bf16)q3.w;
    v16bf b0 = *(const v16bf*)(w1f + ((size_t)(kb * 2 + 0) * 32 + lane) * 16);
    v16bf b1v = *(const v16bf*)(w1f + ((size_t)(kb * 2 + 1) * 32 + lane) * 16);
    acc0 = __builtin_amdgcn_wmma_f32_16x16x32_bf16(false, a, false, b0, (short)0, acc0, false, false);
    acc1 = __builtin_amdgcn_wmma_f32_16x16x32_bf16(false, a, false, b1v, (short)0, acc1, false, false);
  }
  // relu + bias, stash hidden (16x32 per wave) for transposed re-read
  float bb0 = b1[lr], bb1 = b1[16 + lr];
#pragma unroll
  for (int v = 0; v < 8; ++v) {
    int m = half * 8 + v;
    hid[w][m][lr] = (__bf16)fmaxf(acc0[v] + bb0, 0.f);
    hid[w][m][16 + lr] = (__bf16)fmaxf(acc1[v] + bb1, 0.f);
  }
  __syncthreads();
  v16bf a2;
#pragma unroll
  for (int e = 0; e < 16; ++e) a2[e] = hid[w][lr][kmap(e, half)];

  const float* rh = read_head + b * AUXH + rh_off;
  float p[8];
#pragma unroll
  for (int v = 0; v < 8; ++v) p[v] = 0.f;
#pragma unroll
  for (int nt = 0; nt < 4; ++nt) {
    v16bf bf = *(const v16bf*)(w2f + ((size_t)nt * 32 + lane) * 16);
    v8f c = {};
    c = __builtin_amdgcn_wmma_f32_16x16x32_bf16(false, a2, false, bf, (short)0, c, false, false);
    float rv = rh[nt * 16 + lr];
    float bb = b2[nt * 16 + lr];
#pragma unroll
    for (int v = 0; v < 8; ++v) p[v] += rv * sigf(c[v] + bb);
  }
  // reduce the 16 N-columns held across the 16 lanes of each half-group
#pragma unroll
  for (int off = 1; off < 16; off <<= 1) {
#pragma unroll
    for (int v = 0; v < 8; ++v) p[v] += __shfl_xor(p[v], off, 32);
  }
  if (lr == 0) {
    float rh64 = rh[64];
#pragma unroll
    for (int v = 0; v < 8; ++v) {
      int m = m0 + half * 8 + v;
      float uu = u[b * MM + m];
      float g = -logf(1e-20f - logf(1e-20f + uu));
      float lu = last_use[b * MM + m];
      sg_out[b * MM + m] = p[v] + rh64 * sigf(lu) + g;
    }
  }
}

// ---------------------------------------------------------------------------
// Argmax over M, softmax peak value y, gather entry = c * mem[b, argmax, :]
// with c = (1 - y) + y (exact forward value of the straight-through one-hot).
// ---------------------------------------------------------------------------
template <int D>
__global__ void argmax_gather_kernel(const float* __restrict__ sg,
                                     const float* __restrict__ mem,
                                     float* __restrict__ entry) {
  __shared__ float sv[256];
  __shared__ int si[256];
  int b = blockIdx.x, t = threadIdx.x;
  float best = -3.402823e38f;
  int bi = 0;
  for (int m = t; m < MM; m += 256) {
    float v = sg[b * MM + m];
    if (v > best) { best = v; bi = m; }
  }
  sv[t] = best; si[t] = bi;
  __syncthreads();
  for (int s = 128; s > 0; s >>= 1) {
    if (t < s && sv[t + s] > sv[t]) { sv[t] = sv[t + s]; si[t] = si[t + s]; }
    __syncthreads();
  }
  float mx = sv[0];
  int am = si[0];
  __syncthreads();
  float se = 0.f;
  for (int m = t; m < MM; m += 256) se += expf(sg[b * MM + m] - mx);
  sv[t] = se;
  __syncthreads();
  for (int s = 128; s > 0; s >>= 1) {
    if (t < s) sv[t] += sv[t + s];
    __syncthreads();
  }
  float y = 1.f / sv[0];
  float c = (1.f - y) + y;
  const float* src = mem + ((size_t)b * MM + am) * D;
  for (int d = t; d < D; d += 256) entry[b * D + d] = c * src[d];
}

// ---------------------------------------------------------------------------
// Main dual-input GRU via bf16 WMMA.
// wi = Xc(128x512) @ Wih(512x1536); wh = Hc(128x1024) @ Whh(1024x1536)
// r/z accumulators fuse wi+wh; n keeps wi_n / wh_n separate.
// One wave per (mtile 0..7, jtile 0..31); 256 waves total.
// ---------------------------------------------------------------------------
__global__ void __launch_bounds__(256) main_gru_kernel(
    const __bf16* __restrict__ xcf, const __bf16* __restrict__ hcf,
    const __bf16* __restrict__ wihf, const __bf16* __restrict__ whhf,
    const float* __restrict__ bih, const float* __restrict__ bhh,
    const float* __restrict__ h0, float* __restrict__ h1) {
  int tid = threadIdx.x;
  int lane = tid & 31;
  int half = lane >> 4, lr = lane & 15;
  int wv = (blockIdx.x * blockDim.x + tid) >> 5;  // 0..255
  int mt = wv >> 5;                               // 0..7
  int jt = wv & 31;                               // 0..31

  v8f ar = {}, az = {}, ani = {}, anh = {};
#pragma unroll 4
  for (int kb = 0; kb < 16; ++kb) {  // K = 512 (Xc)
    v16bf a = *(const v16bf*)(xcf + ((size_t)(mt * 16 + kb) * 32 + lane) * 16);
    v16bf br = *(const v16bf*)(wihf + ((size_t)(kb * 96 + 0 * 32 + jt) * 32 + lane) * 16);
    v16bf bz = *(const v16bf*)(wihf + ((size_t)(kb * 96 + 1 * 32 + jt) * 32 + lane) * 16);
    v16bf bn = *(const v16bf*)(wihf + ((size_t)(kb * 96 + 2 * 32 + jt) * 32 + lane) * 16);
    ar = __builtin_amdgcn_wmma_f32_16x16x32_bf16(false, a, false, br, (short)0, ar, false, false);
    az = __builtin_amdgcn_wmma_f32_16x16x32_bf16(false, a, false, bz, (short)0, az, false, false);
    ani = __builtin_amdgcn_wmma_f32_16x16x32_bf16(false, a, false, bn, (short)0, ani, false, false);
  }
#pragma unroll 4
  for (int kb = 0; kb < 32; ++kb) {  // K = 1024 (Hc)
    v16bf a = *(const v16bf*)(hcf + ((size_t)(mt * 32 + kb) * 32 + lane) * 16);
    v16bf br = *(const v16bf*)(whhf + ((size_t)(kb * 96 + 0 * 32 + jt) * 32 + lane) * 16);
    v16bf bz = *(const v16bf*)(whhf + ((size_t)(kb * 96 + 1 * 32 + jt) * 32 + lane) * 16);
    v16bf bn = *(const v16bf*)(whhf + ((size_t)(kb * 96 + 2 * 32 + jt) * 32 + lane) * 16);
    ar = __builtin_amdgcn_wmma_f32_16x16x32_bf16(false, a, false, br, (short)0, ar, false, false);
    az = __builtin_amdgcn_wmma_f32_16x16x32_bf16(false, a, false, bz, (short)0, az, false, false);
    anh = __builtin_amdgcn_wmma_f32_16x16x32_bf16(false, a, false, bn, (short)0, anh, false, false);
  }
  int j = jt * 16 + lr;
  float br_ = bih[j] + bhh[j];
  float bz_ = bih[512 + j] + bhh[512 + j];
  float bni = bih[1024 + j];
  float bnh = bhh[1024 + j];
#pragma unroll
  for (int v = 0; v < 8; ++v) {
    int m = mt * 16 + half * 8 + v;
    float r = sigf(ar[v] + br_);
    float z = sigf(az[v] + bz_);
    float n = tanhf((ani[v] + bni) + r * (anh[v] + bnh));
    h1[m * H_DIM + j] = (1.f - z) * n + z * h0[m * H_DIM + j];
  }
}

// ---------------------------------------------------------------------------
extern "C" void kernel_launch(void* const* d_in, const int* in_sizes, int n_in,
                              void* d_out, int out_size, void* d_ws, size_t ws_size,
                              hipStream_t stream) {
  const float* input_    = (const float*)d_in[0];
  const float* h_0       = (const float*)d_in[1];
  const float* aux_h_0   = (const float*)d_in[2];
  const float* imem      = (const float*)d_in[3];
  const float* hmem      = (const float*)d_in[4];
  const float* i_last    = (const float*)d_in[5];
  const float* h_last    = (const float*)d_in[6];
  const float* u_i       = (const float*)d_in[7];
  const float* u_h       = (const float*)d_in[8];
  const float* weight_ih = (const float*)d_in[9];
  const float* weight_hh = (const float*)d_in[10];
  const float* bias_ih   = (const float*)d_in[11];
  const float* bias_hh   = (const float*)d_in[12];
  const float* aux_w_ih  = (const float*)d_in[13];
  const float* aux_w_hh  = (const float*)d_in[14];
  const float* aux_b_ih  = (const float*)d_in[15];
  const float* aux_b_hh  = (const float*)d_in[16];
  const float* ifc_w1    = (const float*)d_in[17];
  const float* ifc_b1    = (const float*)d_in[18];
  const float* ifc_w2    = (const float*)d_in[19];
  const float* ifc_b2    = (const float*)d_in[20];
  const float* hfc_w1    = (const float*)d_in[21];
  const float* hfc_b1    = (const float*)d_in[22];
  const float* hfc_w2    = (const float*)d_in[23];
  const float* hfc_b2    = (const float*)d_in[24];
  // d_in[25]: memcnt == M -> write path does not influence the outputs.

  float* outF = (float*)d_out;
  float* read_head = outF + BB * H_DIM;  // [128,130] second tuple element

  char* ws = (char*)d_ws;
  size_t off = 0;
  auto take = [&](size_t bytes) -> void* {
    void* p = ws + off;
    off = (off + bytes + 255) & ~(size_t)255;
    return p;
  };
  __bf16* w1f_i = (__bf16*)take((size_t)8 * 2 * 512 * 2);       // ifc_w1 frags
  __bf16* w2f_i = (__bf16*)take((size_t)1 * 4 * 512 * 2);       // ifc_w2 frags
  __bf16* w1f_h = (__bf16*)take((size_t)16 * 2 * 512 * 2);      // hfc_w1 frags
  __bf16* w2f_h = (__bf16*)take((size_t)1 * 4 * 512 * 2);       // hfc_w2 frags
  __bf16* wihf  = (__bf16*)take((size_t)16 * 96 * 512 * 2);     // weight_ih frags
  __bf16* whhf  = (__bf16*)take((size_t)32 * 96 * 512 * 2);     // weight_hh frags
  __bf16* xcf   = (__bf16*)take((size_t)8 * 16 * 512 * 2);      // [input|i_entry] frags
  __bf16* hcf   = (__bf16*)take((size_t)8 * 32 * 512 * 2);      // [h0|h_entry] frags
  float* sg_i   = (float*)take((size_t)BB * MM * 4);
  float* sg_h   = (float*)take((size_t)BB * MM * 4);
  float* i_ent  = (float*)take((size_t)BB * I_DIM * 4);
  float* h_ent  = (float*)take((size_t)BB * H_DIM * 4);
  (void)ws_size; (void)in_sizes; (void)n_in; (void)out_size;

  auto blocks = [](int total) { return (total + 255) / 256; };

  // 1) weight prep (cheap, every launch — deterministic)
  prep_bfrag<<<blocks(8 * 2 * 512), 256, 0, stream>>>(ifc_w1, w1f_i, 256, 32);
  prep_bfrag<<<blocks(1 * 4 * 512), 256, 0, stream>>>(ifc_w2, w2f_i, 32, 64);
  prep_bfrag<<<blocks(16 * 2 * 512), 256, 0, stream>>>(hfc_w1, w1f_h, 512, 32);
  prep_bfrag<<<blocks(1 * 4 * 512), 256, 0, stream>>>(hfc_w2, w2f_h, 32, 64);
  prep_bfrag<<<blocks(16 * 96 * 512), 256, 0, stream>>>(weight_ih, wihf, 512, 1536);
  prep_bfrag<<<blocks(32 * 96 * 512), 256, 0, stream>>>(weight_hh, whhf, 1024, 1536);

  // 2) aux GRU -> read_head (written straight into d_out)
  aux_gru_kernel<<<BB, 512, 0, stream>>>(input_, h_0, aux_h_0, aux_w_ih, aux_w_hh,
                                         aux_b_ih, aux_b_hh, read_head);

  // 3) per-slot head FC + scores + gumbel (HBM-bound main work)
  fc_score_kernel<I_DIM><<<dim3(BB, MM / 128), 256, 0, stream>>>(
      imem, i_last, u_i, w1f_i, ifc_b1, w2f_i, ifc_b2, read_head, 0, sg_i);
  fc_score_kernel<H_DIM><<<dim3(BB, MM / 128), 256, 0, stream>>>(
      hmem, h_last, u_h, w1f_h, hfc_b1, w2f_h, hfc_b2, read_head, 65, sg_h);

  // 4) straight-through hard read: argmax + exact ST scale + gather
  argmax_gather_kernel<I_DIM><<<BB, 256, 0, stream>>>(sg_i, imem, i_ent);
  argmax_gather_kernel<H_DIM><<<BB, 256, 0, stream>>>(sg_h, hmem, h_ent);

  // 5) pack GRU activations into A-fragment order (bf16)
  prep_afrag<<<blocks(8 * 16 * 512), 256, 0, stream>>>(input_, i_ent, xcf, 512, 256);
  prep_afrag<<<blocks(8 * 32 * 512), 256, 0, stream>>>(h_0, h_ent, hcf, 1024, 512);

  // 6) main dual-input GRU -> h_1
  main_gru_kernel<<<32, 256, 0, stream>>>(xcf, hcf, wihf, whhf, bias_ih, bias_hh,
                                          h_0, outF);
}
